// PointNetSetAbstraction_22273700397315
// MI455X (gfx1250) — compile-verified, matrix operand order checked by reading, not run on hardware
//
#include <hip/hip_runtime.h>
#include <hip/hip_bf16.h>

#define BB 16
#define NN 256
#define KK 255
#define RTOT (BB*NN*KK)      // 1,044,480 rows
#define NT   (RTOT/16)       // 65,280 M-tiles of 16 rows
#define RAD2 0.04f
#define BNEPS 1e-5f

typedef __attribute__((ext_vector_type(16))) __bf16 v16bf;
typedef __attribute__((ext_vector_type(8)))  float  v8f;

union AFrag { v16bf v; __bf16 e[16]; };
union CAcc  { v8f   v; float  e[8];  };

__device__ inline unsigned short bf16_bits(float f) {
  unsigned u = __float_as_uint(f);
  return (unsigned short)((u + 0x7FFFu + ((u >> 16) & 1u)) >> 16);
}
__device__ inline __bf16 to_bf16(float f) {
  unsigned short s = bf16_bits(f);
  __bf16 b;
  __builtin_memcpy(&b, &s, 2);
  return b;
}

// order-preserving float<->uint encoding for atomicMax/atomicMin
__device__ inline unsigned fenc(float f) {
  unsigned u = __float_as_uint(f);
  return (u & 0x80000000u) ? ~u : (u | 0x80000000u);
}
__device__ inline float fdec(unsigned k) {
  unsigned u = (k & 0x80000000u) ? (k ^ 0x80000000u) : ~k;
  return __uint_as_float(u);
}

__device__ inline void decode_row(int r, int& b, int& n, int& k) {
  b = r / (NN * KK);
  int rem = r - b * (NN * KK);
  n = rem / KK;
  k = rem - n * KK;
}

// ISA 16-bit A-matrix 16x32 striping: element e of v16bf, lane half (0/1)
__device__ inline int kmap(int e, int half) {
  int vg = e >> 1, pos = e & 1;
  int k = (vg < 4) ? (2 * vg + pos) : (16 + 2 * (vg - 4) + pos);
  return k + 8 * half;
}

__device__ inline v8f wmma_bf16(v16bf a, v16bf b, v8f c) {
  return __builtin_amdgcn_wmma_f32_16x16x32_bf16(false, a, false, b, (short)0, c, false, false);
}

// A fragment for K-step s from a row-major bf16 [16][64] tile
// (per lane: two contiguous 16-byte runs -> ds_load_b128 x2)
__device__ inline v16bf load_afrag(const __bf16* A, int lane, int s) {
  AFrag f;
  int m = lane & 15, half = lane >> 4;
#pragma unroll
  for (int e = 0; e < 16; ++e) f.e[e] = A[m * 64 + kmap(e, half) + 32 * s];
  return f.v;
}

// B fragment (K x Nout, LDS layout W[k][n], row stride = stride) — built ONCE, lives in VGPRs
__device__ inline v16bf load_bfrag(const __bf16* W, int stride, int j, int s, int lane) {
  AFrag f;
  int n = (lane & 15) + 16 * j, half = lane >> 4;
#pragma unroll
  for (int e = 0; e < 16; ++e) f.e[e] = W[(kmap(e, half) + 32 * s) * stride + n];
  return f.v;
}

// ---------------- kernels ----------------

__global__ void init_kernel(float* st, int ns, unsigned* mx, unsigned* mn, int nm) {
  int gid = blockIdx.x * blockDim.x + threadIdx.x;
  int str = gridDim.x * blockDim.x;
  for (int i = gid; i < ns; i += str) st[i] = 0.f;
  for (int i = gid; i < nm; i += str) { mx[i] = 0u; mn[i] = 0xFFFFFFFFu; }
}

// ball query: per (b,n), sorted-by-index valid list, backfilled with first valid
__global__ __launch_bounds__(256) void ball_query_kernel(const float* __restrict__ xyz,
                                                         int* __restrict__ idx) {
  int bn = blockIdx.x;
  int b = bn / NN, n = bn % NN;
  int t = threadIdx.x;
  __shared__ int scan[256];
  __shared__ int firstv;
  const float* xb = xyz + (size_t)b * 3 * NN;
  float qx = xb[n], qy = xb[NN + n], qz = xb[2 * NN + n];
  float px = xb[t], py = xb[NN + t], pz = xb[2 * NN + t];
  float dx = px - qx, dy = py - qy, dz = pz - qz;
  float d = dx * dx + dy * dy + dz * dz;
  int flag = (d <= RAD2) ? 1 : 0;
  scan[t] = flag;
  __syncthreads();
  for (int off = 1; off < 256; off <<= 1) {
    int v = (t >= off) ? scan[t - off] : 0;
    __syncthreads();
    scan[t] += v;
    __syncthreads();
  }
  int pos = scan[t] - flag;
  int total = scan[255];
  if (flag && pos == 0) firstv = t;
  __syncthreads();
  int* row = idx + (size_t)bn * KK;
  if (flag && pos < KK) row[pos] = t;
  if (t >= total && t < KK) row[t] = firstv;
}

// Y[b][m][o] = [xyz,pts] @ W0^T ; Xc[b][n][o] = xyz-part only (bias cancels in BN)
__global__ __launch_bounds__(256) void precompute_kernel(const float* __restrict__ xyz,
                                                         const float* __restrict__ pts,
                                                         const float* __restrict__ W0,
                                                         float* __restrict__ Y,
                                                         float* __restrict__ Xc) {
  int i = blockIdx.x * blockDim.x + threadIdx.x;
  if (i >= BB * NN * 64) return;
  int o = i & 63, bm = i >> 6;
  int b = bm / NN, m = bm % NN;
  const float* xb = xyz + (size_t)b * 3 * NN;
  const float* w = W0 + o * 67;
  float sx = w[0] * xb[m] + w[1] * xb[NN + m] + w[2] * xb[2 * NN + m];
  const float* pb = pts + (size_t)b * 64 * NN;
  float sp = 0.f;
#pragma unroll 8
  for (int dch = 0; dch < 64; ++dch) sp += w[3 + dch] * pb[dch * NN + m];
  Xc[(size_t)bm * 64 + o] = sx;
  Y[(size_t)bm * 64 + o] = sx + sp;
}

// pass A: per-channel sum/sumsq of x1 = Y[idx]-Xc (gather, L2-resident)
__global__ __launch_bounds__(256) void stats1_kernel(const float* __restrict__ Y,
                                                     const float* __restrict__ Xc,
                                                     const int* __restrict__ idx,
                                                     float* __restrict__ sum1,
                                                     float* __restrict__ sq1) {
  __shared__ float ls[4][64], lq[4][64];
  int t = threadIdx.x;
  int c = t & 63, g = t >> 6;
  float s = 0.f, q = 0.f;
  int base = blockIdx.x * 512;
  for (int rr = g; rr < 512; rr += 4) {
    int r = base + rr;
    if (r >= RTOT) break;
    int b, n, k;
    decode_row(r, b, n, k);
    int gi = idx[r];
    float v = Y[((size_t)(b * NN + gi)) * 64 + c] - Xc[((size_t)(b * NN + n)) * 64 + c];
    s += v; q += v * v;
  }
  ls[g][c] = s; lq[g][c] = q;
  __syncthreads();
  if (g == 0) {
    s = ls[0][c] + ls[1][c] + ls[2][c] + ls[3][c];
    q = lq[0][c] + lq[1][c] + lq[2][c] + lq[3][c];
    atomicAdd(&sum1[c], s);
    atomicAdd(&sq1[c], q);
  }
}

// stats -> BN affine coefficients a[c], c[c]
__global__ void finalize_kernel(const float* __restrict__ sum, const float* __restrict__ sq,
                                const float* __restrict__ g, const float* __restrict__ beta,
                                float* __restrict__ a, float* __restrict__ c, int C) {
  int t = blockIdx.x * blockDim.x + threadIdx.x;
  if (t >= C) return;
  float mean = sum[t] * (1.f / (float)RTOT);
  float var = sq[t] * (1.f / (float)RTOT) - mean * mean;
  float av = g[t] * rsqrtf(var + BNEPS);
  a[t] = av;
  c[t] = beta[t] - mean * av;
}

// stage one 16x64 bf16 A tile (BN+ReLU applied) with packed 32-bit LDS stores
__device__ inline void stage_atile(__bf16* At, const float* __restrict__ Y,
                                   const float* __restrict__ Xc,
                                   const int* __restrict__ idx, int base, int lane,
                                   const float* sA, const float* sC) {
  int m = lane & 15, cb = lane >> 4;
  int r = base + m;
  int b, n, k;
  decode_row(r, b, n, k);
  int gi = idx[r];
  const float* Yr = Y + ((size_t)(b * NN + gi)) * 64;
  const float* Xr = Xc + ((size_t)(b * NN + n)) * 64;
  unsigned* ArowU = (unsigned*)(At + m * 64);
#pragma unroll
  for (int cc = 0; cc < 32; cc += 2) {
    int ch = cb * 32 + cc;
    float v0 = Yr[ch] - Xr[ch];
    float v1 = Yr[ch + 1] - Xr[ch + 1];
    unsigned lo = bf16_bits(fmaxf(sA[ch] * v0 + sC[ch], 0.f));
    unsigned hi = bf16_bits(fmaxf(sA[ch + 1] * v1 + sC[ch + 1], 0.f));
    ArowU[ch >> 1] = lo | (hi << 16);
  }
}

// pass B: x2 = BN1ReLU(x1) @ W1^T via bf16 WMMA; accumulate stats of x2
__global__ __launch_bounds__(256) void pass2_kernel(const float* __restrict__ Y,
                                                    const float* __restrict__ Xc,
                                                    const int* __restrict__ idx,
                                                    const float* __restrict__ W1,
                                                    const float* __restrict__ a1,
                                                    const float* __restrict__ c1,
                                                    float* __restrict__ sum2,
                                                    float* __restrict__ sq2) {
  __shared__ __bf16 Wl[64 * 64];          // Wl[k][n] = W1[n][k]
  __shared__ __bf16 At[8][16 * 64];       // per-wave A tile
  __shared__ float sA[64], sC[64];
  int t = threadIdx.x;
  for (int i = t; i < 64 * 64; i += 256) {
    int o = i & 63, ch = i >> 6;
    Wl[ch * 64 + o] = to_bf16(W1[o * 64 + ch]);
  }
  if (t < 64) { sA[t] = a1[t]; sC[t] = c1[t]; }
  __syncthreads();

  int lane = t & 31, w = t >> 5;
  // hoist all B fragments into registers: tile-invariant
  v16bf bf[4][2];
#pragma unroll
  for (int j = 0; j < 4; ++j)
#pragma unroll
    for (int s = 0; s < 2; ++s) bf[j][s] = load_bfrag(Wl, 64, j, s, lane);

  int gw = blockIdx.x * 8 + w, nw = gridDim.x * 8;
  float ssum[4] = {0.f, 0.f, 0.f, 0.f}, ssq[4] = {0.f, 0.f, 0.f, 0.f};

  for (int tile = gw; tile < NT; tile += nw) {
    int base = tile * 16;
    stage_atile(&At[w][0], Y, Xc, idx, base, lane, sA, sC);
    asm volatile("s_wait_dscnt 0" ::: "memory");   // wave-local LDS RAW fence

    v16bf af0 = load_afrag(&At[w][0], lane, 0);
    v16bf af1 = load_afrag(&At[w][0], lane, 1);
#pragma unroll
    for (int j = 0; j < 4; ++j) {
      CAcc acc;
#pragma unroll
      for (int v = 0; v < 8; ++v) acc.e[v] = 0.f;
      acc.v = wmma_bf16(af0, bf[j][0], acc.v);
      acc.v = wmma_bf16(af1, bf[j][1], acc.v);
      float s = 0.f, q = 0.f;
#pragma unroll
      for (int v = 0; v < 8; ++v) { s += acc.e[v]; q += acc.e[v] * acc.e[v]; }
      ssum[j] += s; ssq[j] += q;
    }
  }
#pragma unroll
  for (int j = 0; j < 4; ++j) {
    float s = ssum[j] + __shfl_xor(ssum[j], 16, 32);
    float q = ssq[j] + __shfl_xor(ssq[j], 16, 32);
    if (lane < 16) {
      atomicAdd(&sum2[lane + 16 * j], s);
      atomicAdd(&sq2[lane + 16 * j], q);
    }
  }
}

// pass C: recompute x2, BN2ReLU -> f2, x3 = f2 @ W2^T; stats3 + per-(b,n,c) max/min
__global__ __launch_bounds__(256) void pass3_kernel(const float* __restrict__ Y,
                                                    const float* __restrict__ Xc,
                                                    const int* __restrict__ idx,
                                                    const float* __restrict__ W1,
                                                    const float* __restrict__ W2,
                                                    const float* __restrict__ a1,
                                                    const float* __restrict__ c1,
                                                    const float* __restrict__ a2,
                                                    const float* __restrict__ c2,
                                                    float* __restrict__ sum3,
                                                    float* __restrict__ sq3,
                                                    unsigned* __restrict__ gmx,
                                                    unsigned* __restrict__ gmn) {
  __shared__ __bf16 Wl1[64 * 64];
  __shared__ __bf16 Wl2[64 * 128];
  __shared__ __bf16 At[8][16 * 64];
  __shared__ __bf16 Ft[8][16 * 64];
  __shared__ float sA1[64], sC1[64], sA2[64], sC2[64];
  int t = threadIdx.x;
  for (int i = t; i < 64 * 64; i += 256) {
    int o = i & 63, ch = i >> 6;
    Wl1[ch * 64 + o] = to_bf16(W1[o * 64 + ch]);
  }
  for (int i = t; i < 64 * 128; i += 256) {
    int o = i & 127, ch = i >> 7;
    Wl2[ch * 128 + o] = to_bf16(W2[o * 64 + ch]);
  }
  if (t < 64) { sA1[t] = a1[t]; sC1[t] = c1[t]; sA2[t] = a2[t]; sC2[t] = c2[t]; }
  __syncthreads();

  int lane = t & 31, w = t >> 5;
  int half = lane >> 4, ncol = lane & 15;

  // hoist ALL weight fragments into registers (8 + 16 frags, tile-invariant)
  v16bf b1f[4][2], b2f[8][2];
#pragma unroll
  for (int j = 0; j < 4; ++j)
#pragma unroll
    for (int s = 0; s < 2; ++s) b1f[j][s] = load_bfrag(Wl1, 64, j, s, lane);
#pragma unroll
  for (int j = 0; j < 8; ++j)
#pragma unroll
    for (int s = 0; s < 2; ++s) b2f[j][s] = load_bfrag(Wl2, 128, j, s, lane);

  int gw = blockIdx.x * 8 + w, nw = gridDim.x * 8;
  float s3[8] = {0.f, 0.f, 0.f, 0.f, 0.f, 0.f, 0.f, 0.f};
  float q3[8] = {0.f, 0.f, 0.f, 0.f, 0.f, 0.f, 0.f, 0.f};

  for (int tile = gw; tile < NT; tile += nw) {
    int base = tile * 16;
    stage_atile(&At[w][0], Y, Xc, idx, base, lane, sA1, sC1);
    asm volatile("s_wait_dscnt 0" ::: "memory");

    v16bf af0 = load_afrag(&At[w][0], lane, 0);
    v16bf af1 = load_afrag(&At[w][0], lane, 1);
    // layer 2 -> BN2+ReLU -> bf16 Ft tile
#pragma unroll
    for (int j = 0; j < 4; ++j) {
      CAcc acc;
#pragma unroll
      for (int v = 0; v < 8; ++v) acc.e[v] = 0.f;
      acc.v = wmma_bf16(af0, b1f[j][0], acc.v);
      acc.v = wmma_bf16(af1, b1f[j][1], acc.v);
      int ch = ncol + 16 * j;
#pragma unroll
      for (int v = 0; v < 8; ++v) {
        float f = fmaxf(sA2[ch] * acc.e[v] + sC2[ch], 0.f);
        Ft[w][(v + 8 * half) * 64 + ch] = to_bf16(f);
      }
    }
    asm volatile("s_wait_dscnt 0" ::: "memory");

    v16bf ff0 = load_afrag(&Ft[w][0], lane, 0);
    v16bf ff1 = load_afrag(&Ft[w][0], lane, 1);
    // layer 3: 8 N-tiles of 16 -> 128 output channels
#pragma unroll
    for (int j = 0; j < 8; ++j) {
      CAcc acc;
#pragma unroll
      for (int v = 0; v < 8; ++v) acc.e[v] = 0.f;
      acc.v = wmma_bf16(ff0, b2f[j][0], acc.v);
      acc.v = wmma_bf16(ff1, b2f[j][1], acc.v);
      int ch = ncol + 16 * j;
      float mxv = 0.f, mnv = 0.f;
      int lastKey = -1;
#pragma unroll
      for (int v = 0; v < 8; ++v) {
        int r = base + v + 8 * half;
        int b, n, k;
        decode_row(r, b, n, k);
        int key = b * NN + n;
        float val = acc.e[v];
        s3[j] += val;
        q3[j] += val * val;
        if (key != lastKey) {
          if (lastKey >= 0) {
            atomicMax(&gmx[(size_t)lastKey * 128 + ch], fenc(mxv));
            atomicMin(&gmn[(size_t)lastKey * 128 + ch], fenc(mnv));
          }
          lastKey = key;
          mxv = val;
          mnv = val;
        } else {
          mxv = fmaxf(mxv, val);
          mnv = fminf(mnv, val);
        }
      }
      atomicMax(&gmx[(size_t)lastKey * 128 + ch], fenc(mxv));
      atomicMin(&gmn[(size_t)lastKey * 128 + ch], fenc(mnv));
    }
  }
#pragma unroll
  for (int j = 0; j < 8; ++j) {
    float s = s3[j] + __shfl_xor(s3[j], 16, 32);
    float q = q3[j] + __shfl_xor(q3[j], 16, 32);
    if (lane < 16) {
      atomicAdd(&sum3[lane + 16 * j], s);
      atomicAdd(&sq3[lane + 16 * j], q);
    }
  }
}

// BN3+ReLU on extremum (monotone in x for fixed-sign scale), write [B,128,N]
__global__ __launch_bounds__(256) void final_kernel(const unsigned* __restrict__ gmx,
                                                    const unsigned* __restrict__ gmn,
                                                    const float* __restrict__ sum3,
                                                    const float* __restrict__ sq3,
                                                    const float* __restrict__ g2,
                                                    const float* __restrict__ beta2,
                                                    float* __restrict__ out) {
  int i = blockIdx.x * blockDim.x + threadIdx.x;
  if (i >= BB * NN * 128) return;
  int c = i & 127, bn = i >> 7;
  int b = bn / NN, n = bn % NN;
  float mean = sum3[c] * (1.f / (float)RTOT);
  float var = sq3[c] * (1.f / (float)RTOT) - mean * mean;
  float a = g2[c] * rsqrtf(var + BNEPS);
  float cc = beta2[c] - mean * a;
  float v = (a >= 0.f) ? fdec(gmx[i]) : fdec(gmn[i]);
  out[(size_t)b * 128 * NN + (size_t)c * NN + n] = fmaxf(a * v + cc, 0.f);
}

// ---------------- workspace layout ----------------
constexpr size_t OFF_IDX = 0;                                 // RTOT ints
constexpr size_t OFF_Y   = OFF_IDX + (size_t)RTOT * 4;        // 4096*64 f32
constexpr size_t OFF_XC  = OFF_Y + (size_t)BB * NN * 64 * 4;  // 4096*64 f32
constexpr size_t OFF_ST  = OFF_XC + (size_t)BB * NN * 64 * 4; // 768 f32 stats
constexpr size_t OFF_MX  = OFF_ST + 768 * 4;                  // B*N*128 u32
constexpr size_t OFF_MN  = OFF_MX + (size_t)BB * NN * 128 * 4;
constexpr size_t WS_NEED = OFF_MN + (size_t)BB * NN * 128 * 4;

extern "C" void kernel_launch(void* const* d_in, const int* in_sizes, int n_in,
                              void* d_out, int out_size, void* d_ws, size_t ws_size,
                              hipStream_t stream) {
  if (ws_size < WS_NEED) return;
  const float* xyz   = (const float*)d_in[0];
  const float* pts   = (const float*)d_in[1];
  const float* W0    = (const float*)d_in[2];
  const float* g0    = (const float*)d_in[4];
  const float* beta0 = (const float*)d_in[5];
  const float* W1    = (const float*)d_in[6];
  const float* g1    = (const float*)d_in[8];
  const float* beta1 = (const float*)d_in[9];
  const float* W2    = (const float*)d_in[10];
  const float* g2    = (const float*)d_in[12];
  const float* beta2 = (const float*)d_in[13];
  float* out = (float*)d_out;

  char* ws = (char*)d_ws;
  int* idx = (int*)(ws + OFF_IDX);
  float* Y = (float*)(ws + OFF_Y);
  float* Xc = (float*)(ws + OFF_XC);
  float* st = (float*)(ws + OFF_ST);
  float *sum1 = st,       *sq1 = st + 64,  *a1 = st + 128, *c1 = st + 192;
  float *sum2 = st + 256, *sq2 = st + 320, *a2 = st + 384, *c2 = st + 448;
  float *sum3 = st + 512, *sq3 = st + 640;
  unsigned* gmx = (unsigned*)(ws + OFF_MX);
  unsigned* gmn = (unsigned*)(ws + OFF_MN);

  init_kernel<<<256, 256, 0, stream>>>(st, 768, gmx, gmn, BB * NN * 128);
  ball_query_kernel<<<BB * NN, 256, 0, stream>>>(xyz, idx);
  precompute_kernel<<<(BB * NN * 64 + 255) / 256, 256, 0, stream>>>(xyz, pts, W0, Y, Xc);
  stats1_kernel<<<RTOT / 512, 256, 0, stream>>>(Y, Xc, idx, sum1, sq1);
  finalize_kernel<<<1, 64, 0, stream>>>(sum1, sq1, g0, beta0, a1, c1, 64);
  pass2_kernel<<<512, 256, 0, stream>>>(Y, Xc, idx, W1, a1, c1, sum2, sq2);
  finalize_kernel<<<1, 64, 0, stream>>>(sum2, sq2, g1, beta1, a2, c2, 64);
  pass3_kernel<<<512, 256, 0, stream>>>(Y, Xc, idx, W1, W2, a1, c1, a2, c2, sum3, sq3, gmx, gmn);
  final_kernel<<<(BB * NN * 128 + 255) / 256, 256, 0, stream>>>(gmx, gmn, sum3, sq3, g2, beta2, out);
}
